// RivetGNN_74414603371084
// MI455X (gfx1250) — matrix operand order repaired and checked by reading, not compiled, not run on hardware
//
#include <hip/hip_runtime.h>
#include <hip/hip_bf16.h>

#define N_NODES 50000
#define N_EDGES 200000
#define EPS 1e-5f

typedef __attribute__((ext_vector_type(16))) _Float16 v16h;
typedef __attribute__((ext_vector_type(8)))  float    v8f;

// ---------------------------------------------------------------------------
// zero fill
// ---------------------------------------------------------------------------
__global__ void zero_f(float* p, int count) {
    int i = blockIdx.x * blockDim.x + threadIdx.x;
    if (i < count) p[i] = 0.0f;
}

// ---------------------------------------------------------------------------
// Edge kernel: per 16-edge tile (one wave32 each):
//   h = relu(ea @ w1 + b1)          [16 x 16]  (+ h[16]=1 for b2 chunk)
//   msg = (h (x) x[src]) @ W2'      via v_wmma_f32_16x16x32_f16, K=17*32=544
//   atomicAdd msg into agg[dst]
// Block = 128 threads = 4 waves = 4 tiles. 3125 blocks * 4 * 16 = 200000 edges.
// ---------------------------------------------------------------------------
__global__ void __launch_bounds__(128)
edge_conv(const float* __restrict__ x, const int* __restrict__ edge_index,
          const float* __restrict__ edge_attr,
          const float* __restrict__ w1, const float* __restrict__ b1,
          const float* __restrict__ w2, const float* __restrict__ b2,
          float* __restrict__ agg)
{
    // W2'^T staged as f16: T[n*544 + kk], kk = k*32+i for k<16, kk=512+i for b2
    __shared__ __align__(32) _Float16 T[32 * 544];           // 34816 B
    __shared__ float xs[4][16][36];                          // gathered x rows
    __shared__ float hh[4][16][20];                          // edge-MLP hidden
    __shared__ float ea[4][16][8];
    __shared__ int   dsts[4][16];
    __shared__ int   srcs[4][16];

    const int t = threadIdx.x;
    const int w = t >> 5;          // wave id in block
    const int L = t & 31;          // lane id

    // ---- stage W2'^T (transposed so B-fragments are contiguous in K) ----
    for (int q = t; q < 32 * 544; q += 128) {
        int n = q / 544, kk = q - n * 544;
        float v;
        if (kk < 512) { int k = kk >> 5, i = kk & 31; v = w2[k * 1024 + i * 32 + n]; }
        else          { int i = kk - 512;             v = b2[i * 32 + n]; }
        T[q] = (_Float16)v;
    }

    const long tile = (long)blockIdx.x * 4 + w;   // 0..12499
    const long e0   = tile * 16;

    if (L < 16) {
        srcs[w][L] = edge_index[e0 + L];
        dsts[w][L] = edge_index[(long)N_EDGES + e0 + L];
    }
    for (int q = L; q < 128; q += 32) {           // 16 edges x 8 attrs
        int m = q >> 3, i = q & 7;
        ea[w][m][i] = edge_attr[(e0 + m) * 8 + i];
    }
    __syncthreads();   // T, srcs, ea visible

    // ---- gather x[src] rows (16 x 32 f32) via float4 ----
    for (int q = L; q < 128; q += 32) {           // q -> (edge m, float4 qi)
        int m = q >> 3, qi = q & 7;
        const float4 v = ((const float4*)x)[(long)srcs[w][m] * 8 + qi];
        xs[w][m][qi * 4 + 0] = v.x;
        xs[w][m][qi * 4 + 1] = v.y;
        xs[w][m][qi * 4 + 2] = v.z;
        xs[w][m][qi * 4 + 3] = v.w;
    }

    // ---- edge MLP: lanes<16 do outputs 0..7 of edge L, lanes>=16 do 8..15 ----
    {
        int mm = L & 15;
        int ob = (L < 16) ? 0 : 8;
        float er[8];
        #pragma unroll
        for (int i = 0; i < 8; ++i) er[i] = ea[w][mm][i];
        #pragma unroll
        for (int oo = 0; oo < 8; ++oo) {
            int o = ob + oo;
            float acc = b1[o];
            #pragma unroll
            for (int i = 0; i < 8; ++i) acc += er[i] * w1[i * 16 + o];
            hh[w][mm][o] = fmaxf(acc, 0.0f);
        }
        if (L < 16) hh[w][mm][16] = 1.0f;         // multiplier for b2 chunk
    }
    __syncthreads();   // xs, hh visible

    // ---- preload this lane's fixed A-fragment x-values (contiguous k = 16*(L/16) + j) ----
    const int m = L & 15;
    float axs[16];
    #pragma unroll
    for (int j = 0; j < 16; ++j) {
        int K = (L < 16) ? j          // {0..15}
                         : 16 + j;    // {16..31}
        axs[j] = xs[w][m][K];
    }

    // ---- hoist the 17 per-chunk h scalars out of the hot loop ----
    float hv[17];
    #pragma unroll
    for (int kc = 0; kc < 17; ++kc) hv[kc] = hh[w][m][kc];

    v8f c0 = 0.0f, c1 = 0.0f;
    const int nlo  = L & 15;
    const int Koff = (L < 16) ? 0 : 16;
    const _Float16* __restrict__ Tb0 = &T[nlo        * 544 + Koff];
    const _Float16* __restrict__ Tb1 = &T[(nlo + 16) * 544 + Koff];

    // two independent double-buffer sets (even / odd chunks): no register
    // rotation movs, DS refills land 2 chunks ahead of their consumer
    v16h b0a = *(const v16h*)(Tb0);
    v16h b1a = *(const v16h*)(Tb1);
    v16h b0b = *(const v16h*)(Tb0 + 32);
    v16h b1b = *(const v16h*)(Tb1 + 32);

    #pragma unroll 1
    for (int kc = 0; kc < 16; kc += 2) {
        // even chunk: consume buffer A, refill for kc+2 (kc+2 <= 16 always)
        {
            const float h0 = hv[kc];
            v16h a;
            #pragma unroll
            for (int j = 0; j < 16; ++j) a[j] = (_Float16)(h0 * axs[j]);
            c0 = __builtin_amdgcn_wmma_f32_16x16x32_f16(false, a, false, b0a,
                                                        (short)0, c0, false, false);
            c1 = __builtin_amdgcn_wmma_f32_16x16x32_f16(false, a, false, b1a,
                                                        (short)0, c1, false, false);
        }
        b0a = *(const v16h*)(Tb0 + (kc + 2) * 32);
        b1a = *(const v16h*)(Tb1 + (kc + 2) * 32);

        // odd chunk: consume buffer B, refill for kc+3 (clamped; last reload
        // is redundant but keeps the loop branch-free)
        {
            const float h1v = hv[kc + 1];
            v16h a;
            #pragma unroll
            for (int j = 0; j < 16; ++j) a[j] = (_Float16)(h1v * axs[j]);
            c0 = __builtin_amdgcn_wmma_f32_16x16x32_f16(false, a, false, b0b,
                                                        (short)0, c0, false, false);
            c1 = __builtin_amdgcn_wmma_f32_16x16x32_f16(false, a, false, b1b,
                                                        (short)0, c1, false, false);
        }
        const int kn = (kc + 3 < 17) ? (kc + 3) : 16;
        b0b = *(const v16h*)(Tb0 + kn * 32);
        b1b = *(const v16h*)(Tb1 + kn * 32);
    }

    // peeled bias chunk (kc = 16) sits in buffer A
    {
        const float h16 = hv[16];
        v16h a;
        #pragma unroll
        for (int j = 0; j < 16; ++j) a[j] = (_Float16)(h16 * axs[j]);
        c0 = __builtin_amdgcn_wmma_f32_16x16x32_f16(false, a, false, b0a,
                                                    (short)0, c0, false, false);
        c1 = __builtin_amdgcn_wmma_f32_16x16x32_f16(false, a, false, b1a,
                                                    (short)0, c1, false, false);
    }

    // ---- scatter-add to agg[dst] (C layout: VGPR r -> row r + 8*(L>=16)) ----
    const int rb = (L < 16) ? 0 : 8;
    #pragma unroll
    for (int r = 0; r < 8; ++r) {
        int d = dsts[w][rb + r];
        atomicAdd(&agg[(long)d * 32 + nlo],      c0[r]);
        atomicAdd(&agg[(long)d * 32 + nlo + 16], c1[r]);
    }
}

// ---------------------------------------------------------------------------
// Node kernel: y = x@root + agg + bias ; accumulate BN sum/sumsq per channel
// ---------------------------------------------------------------------------
__global__ void __launch_bounds__(256)
node_update(const float* __restrict__ x, const float* __restrict__ agg,
            const float* __restrict__ root, const float* __restrict__ bias,
            float* __restrict__ y, float* __restrict__ stats)
{
    __shared__ float root_s[32 * 32];
    __shared__ float y_s[256][33];
    __shared__ float part[8][64];

    const int t = threadIdx.x;
    for (int q = t; q < 1024; q += 256) root_s[q] = root[q];
    __syncthreads();

    const int n = blockIdx.x * 256 + t;
    float xr[32];
    if (n < N_NODES) {
        #pragma unroll
        for (int i = 0; i < 32; ++i) xr[i] = x[(long)n * 32 + i];
    }
    #pragma unroll 4
    for (int o = 0; o < 32; ++o) {
        float acc = 0.0f;
        if (n < N_NODES) {
            acc = bias[o] + agg[(long)n * 32 + o];
            #pragma unroll
            for (int i = 0; i < 32; ++i) acc += xr[i] * root_s[i * 32 + o];
            y[(long)n * 32 + o] = acc;
        }
        y_s[t][o] = acc;
    }
    __syncthreads();

    // transpose-reduce: 8 groups of 32 rows, per channel
    const int c = t & 31, g = t >> 5;
    float s = 0.0f, s2 = 0.0f;
    for (int j = 0; j < 32; ++j) {
        float v = y_s[g * 32 + j][c];
        s += v; s2 += v * v;
    }
    part[g][c]      = s;
    part[g][32 + c] = s2;
    __syncthreads();
    if (t < 64) {
        float a = 0.0f;
        #pragma unroll
        for (int gg = 0; gg < 8; ++gg) a += part[gg][t];
        atomicAdd(&stats[t], a);   // stats[0..31]=sum, [32..63]=sumsq
    }
}

// ---------------------------------------------------------------------------
// BN(train stats) + ReLU (+ optional residual add)
// ---------------------------------------------------------------------------
__global__ void bn_relu(const float* __restrict__ y, const float* __restrict__ stats,
                        const float* __restrict__ gamma, const float* __restrict__ beta,
                        const float* __restrict__ identity, float* __restrict__ out,
                        int addIdentity)
{
    int idx = blockIdx.x * blockDim.x + threadIdx.x;
    if (idx >= N_NODES * 32) return;
    int c = idx & 31;
    const float inv_n = 1.0f / (float)N_NODES;
    float mu  = stats[c] * inv_n;
    float var = stats[32 + c] * inv_n - mu * mu;
    float v = (y[idx] - mu) * rsqrtf(var + EPS) * gamma[c] + beta[c];
    v = fmaxf(v, 0.0f);
    if (addIdentity) v += identity[idx];
    out[idx] = v;
}

// ---------------------------------------------------------------------------
// Classifier + log_softmax
// ---------------------------------------------------------------------------
__global__ void classify(const float* __restrict__ h, const float* __restrict__ cw,
                         const float* __restrict__ cb, float* __restrict__ out)
{
    int n = blockIdx.x * blockDim.x + threadIdx.x;
    if (n >= N_NODES) return;
    float l0 = cb[0], l1 = cb[1], l2 = cb[2];
    #pragma unroll
    for (int i = 0; i < 32; ++i) {
        float v = h[(long)n * 32 + i];
        l0 += v * cw[i * 3 + 0];
        l1 += v * cw[i * 3 + 1];
        l2 += v * cw[i * 3 + 2];
    }
    float mx  = fmaxf(l0, fmaxf(l1, l2));
    float lse = mx + logf(expf(l0 - mx) + expf(l1 - mx) + expf(l2 - mx));
    out[(long)n * 3 + 0] = l0 - lse;
    out[(long)n * 3 + 1] = l1 - lse;
    out[(long)n * 3 + 2] = l2 - lse;
}

// ---------------------------------------------------------------------------
extern "C" void kernel_launch(void* const* d_in, const int* in_sizes, int n_in,
                              void* d_out, int out_size, void* d_ws, size_t ws_size,
                              hipStream_t stream)
{
    (void)in_sizes; (void)n_in; (void)out_size; (void)ws_size;

    const float* x     = (const float*)d_in[0];
    const int*   eidx  = (const int*)  d_in[1];
    const float* eattr = (const float*)d_in[2];
    const float* cls_w = (const float*)d_in[27];
    const float* cls_b = (const float*)d_in[28];

    const size_t NF = (size_t)N_NODES * 32;        // 1.6M floats
    float* W    = (float*)d_ws;
    float* agg  = W;
    float* y    = W + NF;
    float* h1   = W + 2 * NF;                      // layer1 out = identity
    float* h2   = W + 3 * NF;                      // layer2 out
    float* stats= W + 4 * NF;                      // 64 floats
    float* h3   = h1;                              // layer3 out reuses h1

    const int ZB = 256;
    const int nodeBlocks = (N_NODES + 255) / 256;
    const int elemBlocks = (N_NODES * 32 + 255) / 256;

    const float* layer_in[3]  = { x, h1, h2 };
    float*       layer_out[3] = { h1, h2, h3 };
    const float* ident[3]     = { nullptr, h1, nullptr };
    const int    addId[3]     = { 0, 1, 0 };

    for (int l = 0; l < 3; ++l) {
        const int base = 3 + l * 8;
        const float* w1   = (const float*)d_in[base + 0];
        const float* b1   = (const float*)d_in[base + 1];
        const float* w2   = (const float*)d_in[base + 2];
        const float* b2   = (const float*)d_in[base + 3];
        const float* root = (const float*)d_in[base + 4];
        const float* bias = (const float*)d_in[base + 5];
        const float* gmm  = (const float*)d_in[base + 6];
        const float* bta  = (const float*)d_in[base + 7];

        zero_f<<<(int)((NF + ZB - 1) / ZB), ZB, 0, stream>>>(agg, (int)NF);
        zero_f<<<1, 64, 0, stream>>>(stats, 64);

        edge_conv<<<3125, 128, 0, stream>>>(layer_in[l], eidx, eattr,
                                            w1, b1, w2, b2, agg);
        node_update<<<nodeBlocks, 256, 0, stream>>>(layer_in[l], agg, root, bias,
                                                    y, stats);
        bn_relu<<<elemBlocks, 256, 0, stream>>>(y, stats, gmm, bta,
                                                ident[l], layer_out[l], addId[l]);
    }

    classify<<<nodeBlocks, 256, 0, stream>>>(h3, cls_w, cls_b, (float*)d_out);
}
